// HungarianLoss_46084999086744
// MI455X (gfx1250) — matrix operand order repaired and checked by reading, not compiled, not run on hardware
//
#include <hip/hip_runtime.h>
#include <math.h>

typedef __attribute__((ext_vector_type(16))) _Float16 v16h;
typedef __attribute__((ext_vector_type(8)))  float    v8f;

constexpr int BB = 16;    // batches
constexpr int NN = 256;   // rows/cols
constexpr int DD = 128;   // feature dim
constexpr float FINF = 1e30f;

// ---------------------------------------------------------------------------
// Kernel 1: squared row norms for pred and target (2*B*N rows, 128 f32 each)
// ---------------------------------------------------------------------------
__global__ void __launch_bounds__(256)
norms_kernel(const float* __restrict__ pred, const float* __restrict__ tgt,
             float* __restrict__ pp, float* __restrict__ tt) {
    int idx = blockIdx.x * 256 + threadIdx.x;        // 0 .. 2*B*N-1
    bool isP = idx < BB * NN;
    int r = isP ? idx : idx - BB * NN;
    const float* row = (isP ? pred : tgt) + (size_t)r * DD;
    float s = 0.f;
    #pragma unroll
    for (int d = 0; d < DD; d += 4) {
        float4 x = *(const float4*)(row + d);
        s += x.x * x.x + x.y * x.y + x.z * x.z + x.w * x.w;
    }
    (isP ? pp : tt)[r] = s;
}

// ---------------------------------------------------------------------------
// Kernel 2: cost matrix via WMMA. One wave32 per 16x16 tile of cost[b].
// G = P * T^T  (A = pred tile 16x32 per K-step, B = target rows as 32x16),
// cost = sqrt(max(pp + tt - 2G, 0)).
// ---------------------------------------------------------------------------
__global__ void __launch_bounds__(32)
cost_kernel(const float* __restrict__ pred, const float* __restrict__ tgt,
            const float* __restrict__ pp, const float* __restrict__ tt,
            float* __restrict__ cost) {
    int b   = blockIdx.x >> 8;           // 16 x 16 tiles per batch
    int rem = blockIdx.x & 255;
    int m0  = (rem >> 4) << 4;
    int n0  = (rem & 15) << 4;
    int lane = threadIdx.x;
    int hs   = lane >> 4;                // lane half select
    int l15  = lane & 15;

    const float* prow = pred + ((size_t)b * NN + (m0 + l15)) * DD;
    const float* trow = tgt  + ((size_t)b * NN + (n0 + l15)) * DD;

    const int akb = hs ? 16 : 0;         // A: lanes 0-15 K 0..15; lanes 16-31 K 16..31
    const int bkb = hs ? 16 : 0;         // B: lanes 0-15 K 0..15; lanes 16-31 K 16..31

    v8f c = {};
    #pragma unroll
    for (int kk = 0; kk < 4; ++kk) {     // K = 128 in 4 steps of 32
        int k0 = kk * 32;
        v16h a, bm;
        #pragma unroll
        for (int e = 0; e < 8; ++e) {
            a[e]     = (_Float16)prow[k0 + akb + e];
            a[e + 8] = (_Float16)prow[k0 + akb + 8 + e];
        }
        #pragma unroll
        for (int e = 0; e < 16; ++e)
            bm[e] = (_Float16)trow[k0 + bkb + e];
        c = __builtin_amdgcn_wmma_f32_16x16x32_f16(
                /*neg_a=*/false, a, /*neg_b=*/false, bm,
                /*c_mod=*/(short)0, c, /*reuse_a=*/false, /*reuse_b=*/false);
    }

    int n = n0 + l15;
    float tn = tt[b * NN + n];
    #pragma unroll
    for (int r = 0; r < 8; ++r) {        // C layout: VGPR r -> M = r + 8*hs, N = l15
        int m = m0 + r + hs * 8;
        float d2 = pp[b * NN + m] + tn - 2.0f * c[r];
        cost[((size_t)b * NN + m) * NN + n] = sqrtf(fmaxf(d2, 0.0f));
    }
}

// ---------------------------------------------------------------------------
// Kernel 3: exact Hungarian (JV potentials), one 256-thread block per batch.
// Thread j owns column j+1; argmin over free columns via LDS tree reduction
// with lowest-index tie-break (matches np.argmin). Deterministic.
// ---------------------------------------------------------------------------
__global__ void __launch_bounds__(256)
hungarian_kernel(const float* __restrict__ cost, int* __restrict__ cols) {
    int b = blockIdx.x;
    int tid = threadIdx.x;
    const float* C = cost + (size_t)b * NN * NN;

    __shared__ float u[NN + 1], v[NN + 1], minv[NN + 1];
    __shared__ int   p[NN + 1], way[NN + 1], used[NN + 1];
    __shared__ float sval[NN];
    __shared__ int   sidx[NN];
    __shared__ int   sh_j0, sh_i0, sh_done;

    for (int j = tid; j <= NN; j += NN) { u[j] = 0.f; v[j] = 0.f; p[j] = 0; way[j] = 0; }
    __syncthreads();

    for (int i = 1; i <= NN; ++i) {
        for (int j = tid; j <= NN; j += NN) { minv[j] = FINF; used[j] = 0; }
        if (tid == 0) { p[0] = i; sh_j0 = 0; }
        __syncthreads();

        while (true) {
            if (tid == 0) { used[sh_j0] = 1; sh_i0 = p[sh_j0]; }
            __syncthreads();
            int j0 = sh_j0;
            int i0 = sh_i0;
            int j  = tid + 1;
            float mv = FINF;
            if (!used[j]) {
                float cur = C[(size_t)(i0 - 1) * NN + (j - 1)] - u[i0] - v[j];
                mv = minv[j];
                if (cur < mv) { mv = cur; minv[j] = cur; way[j] = j0; }
            }
            sval[tid] = mv;
            sidx[tid] = j;
            __syncthreads();
            #pragma unroll
            for (int s = 128; s > 0; s >>= 1) {
                if (tid < s) {
                    float v2 = sval[tid + s]; int i2 = sidx[tid + s];
                    float v1 = sval[tid];     int i1 = sidx[tid];
                    if (v2 < v1 || (v2 == v1 && i2 < i1)) { sval[tid] = v2; sidx[tid] = i2; }
                }
                __syncthreads();
            }
            float delta = sval[0];
            int   j1    = sidx[0];
            // u[p[used]] += delta ; v[used] -= delta ; minv[free] -= delta
            for (int jj = tid; jj <= NN; jj += NN) {
                if (used[jj]) { u[p[jj]] += delta; v[jj] -= delta; }
                else          { minv[jj] -= delta; }
            }
            __syncthreads();
            if (tid == 0) { sh_j0 = j1; sh_done = (p[j1] == 0); }
            __syncthreads();
            if (sh_done) break;
        }
        if (tid == 0) {                  // augment along 'way' chain (serial, short)
            int j0 = sh_j0;
            while (j0 != 0) { int j1 = way[j0]; p[j0] = p[j1]; j0 = j1; }
        }
        __syncthreads();
    }
    // col[p[j]-1] = j-1
    cols[b * NN + (p[tid + 1] - 1)] = tid;
}

// ---------------------------------------------------------------------------
// Kernel 4: per-batch MSE + last-channel BCE (deterministic LDS reduction)
// ---------------------------------------------------------------------------
__global__ void __launch_bounds__(256)
loss_kernel(const float* __restrict__ pred, const float* __restrict__ tgt,
            const int* __restrict__ cols, float* __restrict__ lossbuf) {
    int b = blockIdx.x;
    int tid = threadIdx.x;                       // row index
    const float* prow = pred + ((size_t)b * NN + tid) * DD;
    int cj = cols[b * NN + tid];
    const float* trow = tgt + ((size_t)b * NN + cj) * DD;

    float s = 0.f;
    #pragma unroll
    for (int d = 0; d < DD; d += 4) {
        float4 x = *(const float4*)(prow + d);
        float4 y = *(const float4*)(trow + d);
        float dx = x.x - y.x, dy = x.y - y.y, dz = x.z - y.z, dw = x.w - y.w;
        s += dx * dx + dy * dy + dz * dz + dw * dw;
    }
    float pv  = prow[DD - 1];
    float tv  = trow[DD - 1];
    float lp  = fmaxf(logf(pv), -100.f);
    float l1p = fmaxf(log1pf(-pv), -100.f);
    float bce = -(tv * lp + (1.f - tv) * l1p);

    __shared__ float smse[256], sbce[256];
    smse[tid] = s; sbce[tid] = bce;
    __syncthreads();
    #pragma unroll
    for (int st = 128; st > 0; st >>= 1) {
        if (tid < st) { smse[tid] += smse[tid + st]; sbce[tid] += sbce[tid + st]; }
        __syncthreads();
    }
    if (tid == 0) {
        lossbuf[b]      = smse[0] / (float)(NN * DD);
        lossbuf[BB + b] = sbce[0] / (float)NN;
    }
}

__global__ void finalize_kernel(const float* __restrict__ lossbuf, float* __restrict__ out) {
    if (threadIdx.x == 0) {
        float s = 0.f;
        for (int i = 0; i < 2 * BB; ++i) s += lossbuf[i];
        out[0] = s / (float)(2 * BB);
    }
}

// ---------------------------------------------------------------------------
extern "C" void kernel_launch(void* const* d_in, const int* in_sizes, int n_in,
                              void* d_out, int out_size, void* d_ws, size_t ws_size,
                              hipStream_t stream) {
    const float* pred = (const float*)d_in[0];   // (16,256,128) f32
    const float* tgt  = (const float*)d_in[1];   // (16,256,128) f32
    float* out = (float*)d_out;                  // scalar f32

    // workspace layout
    float* cost    = (float*)d_ws;                         // B*N*N
    float* pp      = cost + (size_t)BB * NN * NN;          // B*N
    float* tt      = pp + BB * NN;                         // B*N
    int*   cols    = (int*)(tt + BB * NN);                 // B*N
    float* lossbuf = (float*)(cols + BB * NN);             // 2*B

    norms_kernel<<<(2 * BB * NN) / 256, 256, 0, stream>>>(pred, tgt, pp, tt);
    cost_kernel<<<BB * (NN / 16) * (NN / 16), 32, 0, stream>>>(pred, tgt, pp, tt, cost);
    hungarian_kernel<<<BB, 256, 0, stream>>>(cost, cols);
    loss_kernel<<<BB, 256, 0, stream>>>(pred, tgt, cols, lossbuf);
    finalize_kernel<<<1, 32, 0, stream>>>(lossbuf, out);
}
